// DSAVarlenSparseAttentionOptimized_40372692583234
// MI455X (gfx1250) — matrix operand order repaired and checked by reading, not compiled, not run on hardware
//
#include <hip/hip_runtime.h>
#include <stdint.h>

// DSA varlen sparse attention for MI455X (gfx1250, wave32).
// One wave per block; each block handles one token x 4 heads.
// Gather K/V rows (f32) -> hw-convert to bf16 -> LDS tile -> v_wmma_f32_16x16x32_bf16.

constexpr int H    = 16;   // heads
constexpr int D    = 64;   // head dim
constexpr int TOPK = 64;   // selected keys per token
constexpr int HPB  = 4;    // heads per block

constexpr int ROWSTRIDE  = 144;                    // bytes per bf16 tile row (128 + 16 pad, 16B aligned)
constexpr int TILE_BYTES = TOPK * ROWSTRIDE;       // 9216
constexpr int LDS_BYTES  = TILE_BYTES + 3 * TOPK * 4;  // + GI + scores + weights = 9984

typedef __attribute__((ext_vector_type(16))) __bf16 bf16x16;
typedef __attribute__((ext_vector_type(4)))  __bf16 bf16x4;
typedef __attribute__((ext_vector_type(8)))  float  f32x8;

union BF16Tile { bf16x16 v; __bf16 b[16]; unsigned short s[16]; uint4 q[2]; };

// hardware packed f32 -> bf16 (RNE) for 4 elements
__device__ __forceinline__ uint2 cvt4_bf16(float4 f) {
  bf16x4 b;
  b.x = (__bf16)f.x; b.y = (__bf16)f.y;
  b.z = (__bf16)f.z; b.w = (__bf16)f.w;
  return __builtin_bit_cast(uint2, b);
}

__device__ __forceinline__ f32x8 wmma_bf16(bf16x16 a, bf16x16 b, f32x8 c) {
  // (neg_a, A, neg_b, B, c_mod, C, reuse_a, reuse_b)
  return __builtin_amdgcn_wmma_f32_16x16x32_bf16(false, a, false, b, (short)0, c, false, false);
}

__global__ __launch_bounds__(32) void dsa_sparse_attn_kernel(
    const float* __restrict__ q,  const float* __restrict__ kg,
    const float* __restrict__ vg, const int* __restrict__ cu,
    const int* __restrict__ tki,  const float* __restrict__ tks,
    float* __restrict__ out, int ndocs)
{
  extern __shared__ __align__(16) char smem[];
  char*  TILE = smem;                                     // 64 x 144B bf16 tile (K, then reused for V)
  int*   GI = (int*)  (smem + TILE_BYTES);                // 64 clamped global key ids
  float* SS = (float*)(smem + TILE_BYTES + TOPK * 4);     // 64 raw scores (pre-scaled by 1/sqrt(D))
  float* WW = (float*)(smem + TILE_BYTES + 2 * TOPK * 4); // 64 normalized weights

  const int t    = blockIdx.x;
  const int lane = threadIdx.x & 31;
  const int half = (lane >> 4) & 1;  // 0: lanes 0-15, 1: lanes 16-31
  const int c16  = lane & 15;

  // --- owning document bounds (searchsorted over cu_seqlens) ---
  int seg = 0;
  for (int j = 1; j < ndocs; ++j) seg += (t >= cu[j]) ? 1 : 0;
  const int start = cu[seg];
  const int len   = cu[seg + 1] - start;

  // --- clamp top-k indices into owning document (2 per lane) ---
  {
    int e0 = tki[(size_t)t * TOPK + lane];
    int e1 = tki[(size_t)t * TOPK + lane + 32];
    GI[lane]      = start + min(max(e0 - start, 0), len - 1);
    GI[lane + 32] = start + min(max(e1 - start, 0), len - 1);
  }
  __syncthreads();

  for (int hi = 0; hi < HPB; ++hi) {
    const int h = blockIdx.y * HPB + hi;

    // ---- gather K rows (coalesced: half-wave covers one 256B row), hw cvt f32->bf16, to LDS ----
    #pragma unroll 4
    for (int it = 0; it < 32; ++it) {
      int row = it * 2 + half;
      int g   = GI[row];
      const float4 f = *(const float4*)(kg + (size_t)g * (H * D) + (size_t)h * D + c16 * 4);
      *(uint2*)(TILE + row * ROWSTRIDE + c16 * 8) = cvt4_bf16(f);
    }

    // ---- B(q): q * scale broadcast into all 16 columns; B layout: lane col = lane%16,
    //      K rows = (lane<16 ? {0..7,16..23} : {8..15,24..31}) ----
    const float* qrow = q + ((size_t)t * H + h) * D;
    BF16Tile Bq[2];
    #pragma unroll
    for (int dh = 0; dh < 2; ++dh) {
      const float* qp = qrow + dh * 32 + half * 8;
      #pragma unroll
      for (int j = 0; j < 16; ++j) Bq[dh].b[j] = (__bf16)(qp[(j & 7) + (j >> 3) * 16] * 0.125f);  // fold 1/sqrt(64)
    }
    __syncthreads();

    // ---- QK^T: 4 key chunks x 2 dim-halves of v_wmma_f32_16x16x32_bf16 ----
    // A layout (16-bit 16x32): lanes<16 hold K {0..7,16..23}, lanes>=16 hold {8..15,24..31}
    #pragma unroll
    for (int chunk = 0; chunk < 4; ++chunk) {
      f32x8 acc = {0.f,0.f,0.f,0.f,0.f,0.f,0.f,0.f};
      #pragma unroll
      for (int dh = 0; dh < 2; ++dh) {
        int base = (chunk * 16 + c16) * ROWSTRIDE + dh * 64 + half * 16;
        BF16Tile A;
        A.q[0] = *(const uint4*)(TILE + base);
        A.q[1] = *(const uint4*)(TILE + base + 32);
        acc = wmma_bf16(A.v, Bq[dh].v, acc);
      }
      // C[m,n] identical over n; lane 0 holds s[chunk*16+0..7], lane 16 holds +8..15
      if (c16 == 0) {
        float* dst = SS + chunk * 16 + half * 8;
        dst[0] = acc[0]; dst[1] = acc[1]; dst[2] = acc[2]; dst[3] = acc[3];
        dst[4] = acc[4]; dst[5] = acc[5]; dst[6] = acc[6]; dst[7] = acc[7];
      }
    }
    __syncthreads();

    // ---- fused softmax * selector-score * renorm: w_i = e^{s_i-m} p_i / sum_j e^{s_j-m} p_j ----
    float s0 = SS[lane], s1 = SS[lane + 32];
    float m = fmaxf(s0, s1);
    #pragma unroll
    for (int off = 16; off > 0; off >>= 1) m = fmaxf(m, __shfl_xor(m, off, 32));
    float p0 = tks[(size_t)t * TOPK + lane];
    float p1 = tks[(size_t)t * TOPK + lane + 32];
    float e0 = __expf(s0 - m) * p0;
    float e1 = __expf(s1 - m) * p1;
    float sum = e0 + e1;
    #pragma unroll
    for (int off = 16; off > 0; off >>= 1) sum += __shfl_xor(sum, off, 32);
    float inv = 1.0f / (sum + 1e-12f);
    WW[lane] = e0 * inv; WW[lane + 32] = e1 * inv;
    __syncthreads();

    // ---- A(w): w broadcast into all 16 rows (A layout as above) ----
    BF16Tile Aw[2];
    #pragma unroll
    for (int kb = 0; kb < 2; ++kb) {
      int ko = half * 8;
      #pragma unroll
      for (int j = 0; j < 8; ++j) {
        Aw[kb].b[j]     = (__bf16)WW[kb * 32 + ko + j];
        Aw[kb].b[8 + j] = (__bf16)WW[kb * 32 + 16 + ko + j];
      }
    }

    // ---- gather V rows into the same LDS tile (K tile is dead) ----
    #pragma unroll 4
    for (int it = 0; it < 32; ++it) {
      int row = it * 2 + half;
      int g   = GI[row];
      const float4 f = *(const float4*)(vg + (size_t)g * (H * D) + (size_t)h * D + c16 * 4);
      *(uint2*)(TILE + row * ROWSTRIDE + c16 * 8) = cvt4_bf16(f);
    }
    __syncthreads();

    // ---- P.V: 4 dim chunks x 2 key groups; B[k,n] = V[key k, dchunk+n] read from LDS ----
    const unsigned short* VT = (const unsigned short*)TILE;
    float* outp = out + ((size_t)t * H + h) * D;
    #pragma unroll
    for (int dch = 0; dch < 4; ++dch) {
      f32x8 acc = {0.f,0.f,0.f,0.f,0.f,0.f,0.f,0.f};
      int d = dch * 16 + c16;
      #pragma unroll
      for (int kb = 0; kb < 2; ++kb) {
        BF16Tile Bv;
        int krow = kb * 32 + half * 8;
        #pragma unroll
        for (int j = 0; j < 16; ++j)
          Bv.s[j] = VT[(size_t)(krow + (j & 7) + (j >> 3) * 16) * (ROWSTRIDE / 2) + d];
        acc = wmma_bf16(Aw[kb].v, Bv.v, acc);
      }
      // C[m,n] identical over m; lane n holds O[dchunk+n] in every C register
      if (lane < 16) outp[dch * 16 + lane] = acc[0];
    }
    __syncthreads();  // tile reused by next head
  }
}

extern "C" void kernel_launch(void* const* d_in, const int* in_sizes, int n_in,
                              void* d_out, int out_size, void* d_ws, size_t ws_size,
                              hipStream_t stream) {
  const float* q  = (const float*)d_in[0];
  const float* k  = (const float*)d_in[1];
  const float* v  = (const float*)d_in[2];
  const int*   cu = (const int*)  d_in[3];
  const int*   ti = (const int*)  d_in[4];
  const float* ts = (const float*)d_in[5];
  float* out = (float*)d_out;

  const int T     = in_sizes[0] / (H * D);
  const int ndocs = in_sizes[3] - 1;

  dim3 grid(T, H / HPB);
  dsa_sparse_attn_kernel<<<grid, 32, LDS_BYTES, stream>>>(q, k, v, cu, ti, ts, out, ndocs);
}
